// HybridPDEExtractor_77352361001327
// MI455X (gfx1250) — compile-verified, matrix operand order checked by reading, not run on hardware
//
#include <hip/hip_runtime.h>
#include <cstdint>
#include <cstddef>

// ---------------------------------------------------------------------------
// Problem constants (from reference): B=64, C=3, S=48 -> F=6912, TOT=442368
// ---------------------------------------------------------------------------
#define B_   64
#define C_   3
#define S_   48
#define HW_  (S_ * S_)        // 2304
#define F_   (C_ * HW_)       // 6912
#define TOT_ (B_ * F_)        // 442368

typedef __bf16 bf16;
typedef __attribute__((ext_vector_type(16))) __bf16 v16bf;
typedef __attribute__((ext_vector_type(8)))  float  v8f;

__device__ __forceinline__ bf16 f2bf(float f) {
  union { float f; uint32_t u; } a; a.f = f;
  uint32_t r = (a.u + 0x7FFFu + ((a.u >> 16) & 1u)) >> 16;   // round-nearest-even
  union { unsigned short s; bf16 h; } o; o.s = (unsigned short)r;
  return o.h;
}

// CDNA5 async global->LDS copy (ASYNCcnt-tracked), per CDNA5 ISA
__device__ __forceinline__ void async_load_b128(uint32_t lds_addr, const void* gptr) {
  asm volatile("global_load_async_to_lds_b128 %0, %1, off"
               :: "v"(lds_addr), "v"((uint64_t)(uintptr_t)gptr)
               : "memory");
}
__device__ __forceinline__ void wait_async0() {
  asm volatile("s_wait_asynccnt 0" ::: "memory");
}

// ---------------------------------------------------------------------------
// GEMM: C(64 x N) = A(64 x K) @ Bt(N x K)^T   (bf16 in, fp32 accumulate)
// One workgroup = 8 waves, 128 output columns; wave w -> 16 columns, all 64 rows.
// A chunks (64x64 bf16 = 8KB) double-buffered in LDS via async loads issued a
// chunk ahead of compute; B streamed from global (contiguous rows of Bt).
// Fragment layouts per ISA 7.12.2 (16-bit A 16x32): lane l = m + 16*sel,
// halves[0..7] = K(kk + 8*sel ..), halves[8..15] = K(kk+16+8*sel ..).
// ---------------------------------------------------------------------------
__global__ __launch_bounds__(256)
void gemm_wmma_kernel(const bf16* __restrict__ A, const bf16* __restrict__ Bt,
                      float* __restrict__ Cmat, int Kdim, int Nn) {
  alignas(16) __shared__ bf16 As[2][64 * 64];   // double-buffered 8KB tiles

  const int tid  = threadIdx.x;
  const int wave = tid >> 5;
  const int lane = tid & 31;
  const int sel  = lane >> 4;     // 0: low half-group, 1: high half-group
  const int lr   = lane & 15;
  const int n0   = blockIdx.x * 128 + wave * 16;
  const int n    = n0 + lr;

  v8f acc[4] = {};

  const uint32_t ldsBase = (uint32_t)(uintptr_t)&As[0][0];
  // this thread's two 16B staging assignments (fixed across chunks)
  const int id0 = tid, id1 = tid + 256;
  const int m0s = id0 >> 3, kc0 = (id0 & 7) * 8;
  const int m1s = id1 >> 3, kc1 = (id1 & 7) * 8;

  auto stage = [&](int bsel, int k0) {
    async_load_b128(ldsBase + (uint32_t)(bsel * 8192 + (m0s * 64 + kc0) * 2),
                    A + (size_t)m0s * Kdim + k0 + kc0);
    async_load_b128(ldsBase + (uint32_t)(bsel * 8192 + (m1s * 64 + kc1) * 2),
                    A + (size_t)m1s * Kdim + k0 + kc1);
  };

  // prologue: stage chunk 0 into buffer 0
  stage(0, 0);
  wait_async0();
  __syncthreads();

  int buf = 0;
  for (int k0 = 0; k0 < Kdim; k0 += 64) {
    // issue async stage of next chunk into the other buffer (overlaps compute)
    if (k0 + 64 < Kdim) stage(buf ^ 1, k0 + 64);
    __builtin_prefetch(Bt + (size_t)n * Kdim + k0 + 64, 0, 1);

    const bf16* asBase = &As[buf][0];
#pragma unroll
    for (int kk = 0; kk < 64; kk += 32) {
      union U { int4 q[2]; v16bf v; };
      // B fragment: lane holds column n of B == row n of Bt (contiguous in k)
      const bf16* bp = Bt + (size_t)n * Kdim + k0 + kk + 8 * sel;
      U bu;
      bu.q[0] = *(const int4*)bp;
      bu.q[1] = *(const int4*)(bp + 16);

      // batch all four A fragments first so ds_loads clause together
      U au[4];
#pragma unroll
      for (int mt = 0; mt < 4; ++mt) {
        const bf16* ap = asBase + (mt * 16 + lr) * 64 + kk + 8 * sel;
        au[mt].q[0] = *(const int4*)ap;        // ds_load_b128
        au[mt].q[1] = *(const int4*)(ap + 16);
      }
#pragma unroll
      for (int mt = 0; mt < 4; ++mt)
        acc[mt] = __builtin_amdgcn_wmma_f32_16x16x32_bf16(
            false, au[mt].v, false, bu.v, (short)0, acc[mt], false, false);
    }

    // next chunk's async writes must be visible to all waves before next iter
    wait_async0();
    __syncthreads();
    buf ^= 1;
  }

  // C/D layout: VGPR r -> row (r + 8*sel), col (lane & 15)
#pragma unroll
  for (int mt = 0; mt < 4; ++mt)
#pragma unroll
    for (int r = 0; r < 8; ++r)
      Cmat[(size_t)(mt * 16 + r + 8 * sel) * Nn + n] = acc[mt][r];
}

// ---------------------------------------------------------------------------
// K-matrix prep: fp32 -> bf16 straight copy + bf16 transpose (for S@K GEMMs)
// ---------------------------------------------------------------------------
__global__ void kprep_kernel(const float* __restrict__ Kf,
                             bf16* __restrict__ Kb, bf16* __restrict__ KTb) {
  __shared__ bf16 tile[32][33];
  const int nbx = F_ / 32;
  int bx = blockIdx.x % nbx, by = blockIdx.x / nbx;
  int r0 = by * 32, c0 = bx * 32;
  int tr = threadIdx.x >> 5, tc = threadIdx.x & 31;
#pragma unroll
  for (int p = 0; p < 4; ++p) {
    int r = p * 8 + tr;
    bf16 h = f2bf(Kf[(size_t)(r0 + r) * F_ + c0 + tc]);
    Kb[(size_t)(r0 + r) * F_ + c0 + tc] = h;
    tile[r][tc] = h;
  }
  __syncthreads();
#pragma unroll
  for (int p = 0; p < 4; ++p) {
    int r = p * 8 + tr;
    KTb[(size_t)(c0 + r) * F_ + r0 + tc] = tile[tc][r];
  }
}

// ---------------------------------------------------------------------------
// Diffusion: channel mix + Thomas tridiagonal solves (W then H direction)
// ---------------------------------------------------------------------------
__global__ void mix_kernel(const float* __restrict__ in, const float* __restrict__ mixm,
                           float* __restrict__ out) {
  int i = blockIdx.x * blockDim.x + threadIdx.x;   // over B*HW
  if (i >= B_ * HW_) return;
  int b = i / HW_, hw = i % HW_;
  const float* ib = in + (size_t)b * C_ * HW_ + hw;
  float u0 = ib[0], u1 = ib[HW_], u2 = ib[2 * HW_];
  float* ob = out + (size_t)b * C_ * HW_ + hw;
  ob[0]        = mixm[0] * u0 + mixm[1] * u1 + mixm[2] * u2;
  ob[HW_]      = mixm[3] * u0 + mixm[4] * u1 + mixm[5] * u2;
  ob[2 * HW_]  = mixm[6] * u0 + mixm[7] * u1 + mixm[8] * u2;
}

__global__ void thomasX_kernel(const float* __restrict__ in, float* __restrict__ out,
                               const float* __restrict__ ab, const float* __restrict__ atc,
                               float t, float half_dt) {
  __shared__ float cs[64][S_ + 1];
  __shared__ float dsx[64][S_ + 1];
  int row = blockIdx.x * 64 + threadIdx.x;         // (b,c,h) rows
  if (row >= B_ * C_ * S_) return;
  int ch = row % (C_ * S_);
  const float* d   = in  + (size_t)row * S_;
  const float* abp = ab  + (size_t)ch * S_;
  const float* atp = atc + (size_t)ch * S_;
  int tl = threadIdx.x;
  float cp = 0.f, dp = 0.f;
  for (int i = 0; i < S_; ++i) {
    float alpha = fminf(fmaxf(abp[i] + atp[i] * t, 1e-6f), 10.f);
    float co = alpha * half_dt;
    float bi = (i == 0 || i == S_ - 1) ? (1.f + co) : (1.f + 2.f * co);
    float ai = (i == 0) ? 0.f : -co;
    float inv = 1.f / (bi - ai * cp + 1e-6f);
    float csv = -co * inv;
    float dsv = (d[i] - ai * dp) * inv;
    cs[tl][i] = csv; dsx[tl][i] = dsv;
    cp = csv; dp = dsv;
  }
  float x = dsx[tl][S_ - 1];
  out[(size_t)row * S_ + S_ - 1] = x;
  for (int i = S_ - 2; i >= 0; --i) {
    x = dsx[tl][i] - cs[tl][i] * x;
    out[(size_t)row * S_ + i] = x;
  }
}

__global__ void thomasY_kernel(float* __restrict__ u,
                               const float* __restrict__ bb, const float* __restrict__ btc,
                               float t, float half_dt) {
  __shared__ float cs[64][S_ + 1];
  __shared__ float dsx[64][S_ + 1];
  int row = blockIdx.x * 64 + threadIdx.x;         // (b,c,w) columns
  if (row >= B_ * C_ * S_) return;
  int c = (row / S_) % C_;
  int w = row % S_;
  int bc = row / S_;                               // b*C + c
  float* base = u + (size_t)bc * HW_ + w;          // element i at base[i*S_]
  int tl = threadIdx.x;
  float cp = 0.f, dp = 0.f;
  for (int i = 0; i < S_; ++i) {
    size_t pidx = (size_t)c * HW_ + (size_t)i * S_ + w;
    float beta = fminf(fmaxf(bb[pidx] + btc[pidx] * t, 1e-6f), 10.f);
    float co = beta * half_dt;
    float bi = (i == 0 || i == S_ - 1) ? (1.f + co) : (1.f + 2.f * co);
    float ai = (i == 0) ? 0.f : -co;
    float inv = 1.f / (bi - ai * cp + 1e-6f);
    float csv = -co * inv;
    float dsv = (base[(size_t)i * S_] - ai * dp) * inv;
    cs[tl][i] = csv; dsx[tl][i] = dsv;
    cp = csv; dp = dsv;
  }
  float x = dsx[tl][S_ - 1];
  base[(size_t)(S_ - 1) * S_] = x;
  for (int i = S_ - 2; i >= 0; --i) {
    x = dsx[tl][i] - cs[tl][i] * x;
    base[(size_t)i * S_] = x;
  }
}

// ---------------------------------------------------------------------------
// symmetric() helpers: f32->bf16, column BN stats, BN+relu->bf16, axpy
// ---------------------------------------------------------------------------
__global__ void cvt_kernel(const float* __restrict__ in, bf16* __restrict__ out, int n) {
  int i = blockIdx.x * blockDim.x + threadIdx.x;
  if (i < n) out[i] = f2bf(in[i]);
}

__global__ void colstats_kernel(const float* __restrict__ X,
                                float* __restrict__ mean, float* __restrict__ var) {
  int j = blockIdx.x * blockDim.x + threadIdx.x;
  if (j >= F_) return;
  float s = 0.f, s2 = 0.f;
  for (int m = 0; m < B_; ++m) {
    float v = X[(size_t)m * F_ + j];
    s += v; s2 += v * v;
  }
  float mu = s * (1.f / B_);
  mean[j] = mu;
  var[j]  = s2 * (1.f / B_) - mu * mu;
}

__global__ void bnrelu_kernel(const float* __restrict__ X,
                              const float* __restrict__ mean, const float* __restrict__ var,
                              const float* __restrict__ g, const float* __restrict__ b,
                              bf16* __restrict__ out) {
  int i = blockIdx.x * blockDim.x + threadIdx.x;
  if (i >= TOT_) return;
  int j = i % F_;
  float v = (X[i] - mean[j]) * rsqrtf(var[j] + 1e-5f) * g[j] + b[j];
  out[i] = f2bf(fmaxf(v, 0.f));
}

__global__ void axpy_kernel(float* __restrict__ dst, const float* __restrict__ src,
                            float s, int n) {
  int i = blockIdx.x * blockDim.x + threadIdx.x;
  if (i < n) dst[i] += s * src[i];
}

__global__ void fill_kernel(float* __restrict__ p, float v, int n) {
  int i = blockIdx.x * blockDim.x + threadIdx.x;
  if (i < n) p[i] = v;
}

// ---------------------------------------------------------------------------
// Final combine (softmax weights) + per-channel BN
// ---------------------------------------------------------------------------
__global__ void combine_kernel(const float* __restrict__ d1, const float* __restrict__ d2,
                               const float* __restrict__ yp, const float* __restrict__ yh,
                               const float* __restrict__ cw, float* __restrict__ out) {
  int i = blockIdx.x * blockDim.x + threadIdx.x;
  if (i >= TOT_) return;
  float c0 = cw[0], c1 = cw[1], c2 = cw[2], c3 = cw[3];
  float mx = fmaxf(fmaxf(c0, c1), fmaxf(c2, c3));
  float e0 = __expf(c0 - mx), e1 = __expf(c1 - mx), e2 = __expf(c2 - mx), e3 = __expf(c3 - mx);
  float inv = 1.f / (e0 + e1 + e2 + e3);
  out[i] = (e0 * d1[i] + e1 * d2[i] + e2 * yp[i] + e3 * yh[i]) * inv;
}

__global__ void chanstats_kernel(const float* __restrict__ x, float* __restrict__ stats) {
  __shared__ float ss[256], ss2[256];
  int c = blockIdx.x;
  float s = 0.f, s2 = 0.f;
  for (int i = threadIdx.x; i < B_ * HW_; i += 256) {
    int b = i / HW_, hw = i % HW_;
    float v = x[((size_t)b * C_ + c) * HW_ + hw];
    s += v; s2 += v * v;
  }
  ss[threadIdx.x] = s; ss2[threadIdx.x] = s2;
  __syncthreads();
  for (int o = 128; o > 0; o >>= 1) {
    if (threadIdx.x < o) { ss[threadIdx.x] += ss[threadIdx.x + o]; ss2[threadIdx.x] += ss2[threadIdx.x + o]; }
    __syncthreads();
  }
  if (threadIdx.x == 0) {
    float n = (float)(B_ * HW_);
    float mu = ss[0] / n;
    stats[c] = mu;
    stats[3 + c] = ss2[0] / n - mu * mu;
  }
}

__global__ void channorm_kernel(float* __restrict__ x, const float* __restrict__ stats,
                                const float* __restrict__ fng, const float* __restrict__ fnb) {
  int i = blockIdx.x * blockDim.x + threadIdx.x;
  if (i >= TOT_) return;
  int c = (i / HW_) % C_;
  x[i] = (x[i] - stats[c]) * rsqrtf(stats[3 + c] + 1e-5f) * fng[c] + fnb[c];
}

// ---------------------------------------------------------------------------
// Host orchestration
// ---------------------------------------------------------------------------
namespace {

struct Scratch {
  bf16 *Kb[3], *KTb[3];
  float *SB, *PB, *utmp, *Zh;
  bf16 *Ybf, *Sbf;
  float *colmean, *colvar, *cstats;
};

inline void run_diffusion(const float* x, float* u, float* utmp,
                          const float* ab, const float* bb,
                          const float* atc, const float* btc,
                          const float* mixm, double dt, int steps, hipStream_t s) {
  hipMemcpyAsync(u, x, (size_t)TOT_ * 4, hipMemcpyDeviceToDevice, s);
  double t = 0.0;
  const float hdt = (float)(dt * 0.5);
  for (int k = 0; k < steps; ++k) {
    mix_kernel<<<(B_ * HW_ + 255) / 256, 256, 0, s>>>(u, mixm, utmp);
    thomasX_kernel<<<(B_ * C_ * S_) / 64, 64, 0, s>>>(utmp, u, ab, atc, (float)t, hdt);
    t += dt * 0.5;
    thomasY_kernel<<<(B_ * C_ * S_) / 64, 64, 0, s>>>(u, bb, btc, (float)t, hdt);
    t += dt * 0.5;
  }
}

// target += scale * (S @ K), where S = relu(BN(Yin @ K^T)); i.e. target += -scale*symmetric(Yin)
inline void run_symmetric(const float* Yin, const bf16* Kb, const bf16* KTb,
                          const float* g, const float* b, float* target, float scale,
                          const Scratch& w, hipStream_t s) {
  const int g256 = (TOT_ + 255) / 256;
  cvt_kernel<<<g256, 256, 0, s>>>(Yin, w.Ybf, TOT_);
  gemm_wmma_kernel<<<F_ / 128, 256, 0, s>>>(w.Ybf, Kb, w.SB, F_, F_);   // KY = Y @ K^T
  colstats_kernel<<<(F_ + 255) / 256, 256, 0, s>>>(w.SB, w.colmean, w.colvar);
  bnrelu_kernel<<<g256, 256, 0, s>>>(w.SB, w.colmean, w.colvar, g, b, w.Sbf);
  gemm_wmma_kernel<<<F_ / 128, 256, 0, s>>>(w.Sbf, KTb, w.PB, F_, F_);  // P = S @ K
  axpy_kernel<<<g256, 256, 0, s>>>(target, w.PB, scale, TOT_);
}

} // namespace

extern "C" void kernel_launch(void* const* d_in, const int* in_sizes, int n_in,
                              void* d_out, int out_size, void* d_ws, size_t ws_size,
                              hipStream_t stream) {
  (void)in_sizes; (void)n_in; (void)out_size; (void)ws_size;

  const float* x    = (const float*)d_in[0];
  const float* ab1  = (const float*)d_in[1];
  const float* bb1  = (const float*)d_in[2];
  const float* atc1 = (const float*)d_in[3];
  const float* btc1 = (const float*)d_in[4];
  const float* mix1 = (const float*)d_in[5];
  const float* ab2  = (const float*)d_in[6];
  const float* bb2  = (const float*)d_in[7];
  const float* atc2 = (const float*)d_in[8];
  const float* btc2 = (const float*)d_in[9];
  const float* mix2 = (const float*)d_in[10];
  const float* Kp   = (const float*)d_in[11];
  const float* gp   = (const float*)d_in[12];
  const float* bp   = (const float*)d_in[13];
  const float* Ky   = (const float*)d_in[14];
  const float* gy   = (const float*)d_in[15];
  const float* by_  = (const float*)d_in[16];
  const float* Kz   = (const float*)d_in[17];
  const float* gz   = (const float*)d_in[18];
  const float* bz   = (const float*)d_in[19];
  const float* cw   = (const float*)d_in[20];
  const float* fng  = (const float*)d_in[21];
  const float* fnb  = (const float*)d_in[22];

  // Output layout: [combined, d1, d2, Yp, Yh], each TOT_ floats
  float* out      = (float*)d_out;
  float* combined = out;
  float* d1 = out + (size_t)TOT_;
  float* d2 = out + (size_t)2 * TOT_;
  float* Yp = out + (size_t)3 * TOT_;
  float* Yh = out + (size_t)4 * TOT_;

  // Workspace carve-up (256B aligned slices)
  char* wsb = (char*)d_ws;
  size_t off = 0;
  auto take = [&](size_t bytes) -> void* {
    void* p = wsb + off;
    off += (bytes + 255) & ~(size_t)255;
    return p;
  };
  const size_t KBYTES = (size_t)F_ * F_ * sizeof(bf16);
  Scratch w;
  for (int i = 0; i < 3; ++i) w.Kb[i]  = (bf16*)take(KBYTES);
  for (int i = 0; i < 3; ++i) w.KTb[i] = (bf16*)take(KBYTES);
  w.SB      = (float*)take((size_t)TOT_ * 4);
  w.PB      = (float*)take((size_t)TOT_ * 4);
  w.utmp    = (float*)take((size_t)TOT_ * 4);
  w.Zh      = (float*)take((size_t)TOT_ * 4);
  w.Ybf     = (bf16*)take((size_t)TOT_ * 2);
  w.Sbf     = (bf16*)take((size_t)TOT_ * 2);
  w.colmean = (float*)take((size_t)F_ * 4);
  w.colvar  = (float*)take((size_t)F_ * 4);
  w.cstats  = (float*)take(64);

  const int g256 = (TOT_ + 255) / 256;
  const int kgrid = (F_ / 32) * (F_ / 32);

  // ---- 1) K matrices -> bf16 (straight + transposed) -----------------------
  kprep_kernel<<<kgrid, 256, 0, stream>>>(Kp, w.Kb[0], w.KTb[0]);
  kprep_kernel<<<kgrid, 256, 0, stream>>>(Ky, w.Kb[1], w.KTb[1]);
  kprep_kernel<<<kgrid, 256, 0, stream>>>(Kz, w.Kb[2], w.KTb[2]);

  // ---- 2) Diffusion layers --------------------------------------------------
  run_diffusion(x, d1, w.utmp, ab1, bb1, atc1, btc1, mix1, 0.001, 8, stream);
  run_diffusion(x, d2, w.utmp, ab2, bb2, atc2, btc2, mix2, 0.002, 5, stream);

  // ---- 3) Parabolic block: Yp += 0.5 * symmetric(Yp) -> Yp -= 0.5 * P ------
  hipMemcpyAsync(Yp, x, (size_t)TOT_ * 4, hipMemcpyDeviceToDevice, stream);
  for (int k = 0; k < 4; ++k)
    run_symmetric(Yp, w.Kb[0], w.KTb[0], gp, bp, Yp, -0.5f, w, stream);

  // ---- 4) Hamiltonian block -------------------------------------------------
  // Yh += 0.8 * (-symmetric(Zh, Ky)) = Yh + 0.8 * P(Zh)
  // Zh -= 0.8 * symmetric(Yh, Kz)    = Zh + 0.8 * P(Yh)
  hipMemcpyAsync(Yh, x, (size_t)TOT_ * 4, hipMemcpyDeviceToDevice, stream);
  fill_kernel<<<g256, 256, 0, stream>>>(w.Zh, 0.f, TOT_);
  for (int k = 0; k < 3; ++k) {
    run_symmetric(w.Zh, w.Kb[1], w.KTb[1], gy, by_, Yh, +0.8f, w, stream);
    run_symmetric(Yh,  w.Kb[2], w.KTb[2], gz, bz,  w.Zh, +0.8f, w, stream);
  }

  // ---- 5) Combine + channel BN ---------------------------------------------
  combine_kernel<<<g256, 256, 0, stream>>>(d1, d2, Yp, Yh, cw, combined);
  chanstats_kernel<<<C_, 256, 0, stream>>>(combined, w.cstats);
  channorm_kernel<<<g256, 256, 0, stream>>>(combined, w.cstats, fng, fnb);
}
